// GatedGCNLayer_64441689309912
// MI455X (gfx1250) — matrix operand order repaired and checked by reading, not compiled, not run on hardware
//
#include <hip/hip_runtime.h>
#include <hip/hip_bf16.h>

typedef __attribute__((ext_vector_type(16))) _Float16 v16h;
typedef __attribute__((ext_vector_type(8)))  float    v8f;

#define DIM      128
#define NN_NODES 50000
#define NE_EDGES 800000

// ---------------------------------------------------------------------------
// Convert one 128x128 f32 weight matrix to f16, stored FRAGMENT-MAJOR so a
// lane's whole 16-half WMMA B-fragment is contiguous (one 32B LDS read):
//   dst index = ((nt*4 + ks)*32 + lane)*16 + j
//   maps to    W[k][n],  n = nt*16 + (lane&15),  k = ks*32 + (lane>>4)*16 + j
// ---------------------------------------------------------------------------
__global__ void cvt_w_swizzle_kernel(const float* __restrict__ w, _Float16* __restrict__ wf) {
    int idx = blockIdx.x * 256 + threadIdx.x;
    if (idx < DIM * DIM) {
        int j    = idx & 15;
        int lane = (idx >> 4) & 31;
        int ks   = (idx >> 9) & 3;
        int nt   = idx >> 11;
        int n = nt * 16 + (lane & 15);
        int k = ks * 32 + (lane >> 4) * 16 + j;
        wf[idx] = (_Float16)w[k * DIM + n];
    }
}

// ---------------------------------------------------------------------------
// Y = X @ W + bias   (X: [Mrows x 128] f32, W: f16 fragment-major, Y f32)
// Block: 256 threads = 8 waves; tile 128 rows x 128 cols.
// Wave w owns rows [w*16, w*16+16); 8 accumulators cover the 8 N-tiles.
// K-loop: 4 steps of 32 with v_wmma_f32_16x16x32_f16; B-frags are single
// 32B LDS vector loads; A-tile global loads are software-pipelined.
// ---------------------------------------------------------------------------
__global__ __launch_bounds__(256)
void gemm_wmma_kernel(const float* __restrict__ X, const _Float16* __restrict__ Wf,
                      const float* __restrict__ bias, float* __restrict__ Y, int Mrows) {
    __shared__ _Float16 sW[DIM * DIM];   // 32 KB fragment-major weight panel

    int tid = threadIdx.x;
    // cooperative 128-bit staging of the (already swizzled) panel into LDS
    {
        const uint4* srcp = (const uint4*)Wf;
        uint4*       dstp = (uint4*)sW;
        #pragma unroll
        for (int i = tid; i < (DIM * DIM) / 8; i += 256) dstp[i] = srcp[i];
    }
    __syncthreads();

    int wave = tid >> 5;
    int lane = tid & 31;
    int lo   = lane & 15;
    int hi   = lane >> 4;
    int laneoff = hi * 16;

    int rowBase = blockIdx.x * 128 + wave * 16;
    int arow = rowBase + lo;
    if (arow >= Mrows) arow = Mrows - 1;           // clamp; OOB rows never stored
    const float* xr = X + (size_t)arow * DIM;

    v8f acc[8] = {};

    // prologue: A-tile loads for K-step 0
    float4 fa0 = *(const float4*)(xr + laneoff);
    float4 fa1 = *(const float4*)(xr + laneoff + 4);
    float4 fa2 = *(const float4*)(xr + laneoff + 8);
    float4 fa3 = *(const float4*)(xr + laneoff + 12);

    #pragma unroll
    for (int ks = 0; ks < 4; ++ks) {
        // issue next K-step's A loads before consuming this step's
        float4 fn0, fn1, fn2, fn3;
        if (ks < 3) {
            int kb = (ks + 1) * 32 + laneoff;
            fn0 = *(const float4*)(xr + kb);
            fn1 = *(const float4*)(xr + kb + 4);
            fn2 = *(const float4*)(xr + kb + 8);
            fn3 = *(const float4*)(xr + kb + 12);
        }

        // B fragments: one aligned 32B LDS vector load each
        v16h bfrag[8];
        #pragma unroll
        for (int nt = 0; nt < 8; ++nt)
            bfrag[nt] = *(const v16h*)(sW + ((size_t)(nt * 4 + ks) * 32 + lane) * 16);

        // A fragment: 16x32 f16, contiguous k = 16*hi + s (same order as B panel)
        v16h a;
        a[0]  = (_Float16)fa0.x; a[1]  = (_Float16)fa0.y; a[2]  = (_Float16)fa0.z; a[3]  = (_Float16)fa0.w;
        a[4]  = (_Float16)fa1.x; a[5]  = (_Float16)fa1.y; a[6]  = (_Float16)fa1.z; a[7]  = (_Float16)fa1.w;
        a[8]  = (_Float16)fa2.x; a[9]  = (_Float16)fa2.y; a[10] = (_Float16)fa2.z; a[11] = (_Float16)fa2.w;
        a[12] = (_Float16)fa3.x; a[13] = (_Float16)fa3.y; a[14] = (_Float16)fa3.z; a[15] = (_Float16)fa3.w;

        #pragma unroll
        for (int nt = 0; nt < 8; ++nt)
            acc[nt] = __builtin_amdgcn_wmma_f32_16x16x32_f16(
                false, a, false, bfrag[nt], (short)0, acc[nt], false, false);

        fa0 = fn0; fa1 = fn1; fa2 = fn2; fa3 = fn3;
    }

    // D layout: VGPR r, lanes 0-15 -> M=r, lanes 16-31 -> M=r+8, N=lane%16.
    #pragma unroll
    for (int nt = 0; nt < 8; ++nt) {
        int col  = nt * 16 + lo;
        float bc = bias[col];
        #pragma unroll
        for (int r = 0; r < 8; ++r) {
            int orow = rowBase + hi * 8 + r;
            if (orow < Mrows) Y[(size_t)orow * DIM + col] = acc[nt][r] + bc;
        }
    }
}

// ---------------------------------------------------------------------------
// Edge stage: e_ij = Ce + Dh[src] + Eh[dst]; sigma = sigmoid(e_ij)
//   num[dst] += sigma*Bh[src];  den[dst] += sigma  (f32 global atomics)
//   per-feature BN partials for e (LDS reduce, then one global atomic each)
// Wave-per-edge; lane owns 4 consecutive features (float4).
// ---------------------------------------------------------------------------
__global__ __launch_bounds__(256)
void edge_kernel(const float* __restrict__ Ce, const float* __restrict__ Dh,
                 const float* __restrict__ Eh, const float* __restrict__ Bh,
                 const int* __restrict__ src, const int* __restrict__ dst,
                 float* __restrict__ eij, float* __restrict__ num, float* __restrict__ den,
                 float* __restrict__ e_sum, float* __restrict__ e_sq) {
    __shared__ float sSum[DIM];
    __shared__ float sSq[DIM];
    int tid = threadIdx.x;
    for (int i = tid; i < DIM; i += 256) { sSum[i] = 0.f; sSq[i] = 0.f; }
    __syncthreads();

    int wave = tid >> 5, lane = tid & 31;
    int f = lane * 4;
    long gw     = (long)blockIdx.x * 8 + wave;
    long stride = (long)gridDim.x * 8;

    float ls0 = 0.f, ls1 = 0.f, ls2 = 0.f, ls3 = 0.f;
    float lq0 = 0.f, lq1 = 0.f, lq2 = 0.f, lq3 = 0.f;

    for (long e = gw; e < NE_EDGES; e += stride) {
        int s = src[e], d = dst[e];
        float4 ce = *(const float4*)(Ce + (size_t)e * DIM + f);
        float4 dh = *(const float4*)(Dh + (size_t)s * DIM + f);
        float4 eh = *(const float4*)(Eh + (size_t)d * DIM + f);
        float4 bh = *(const float4*)(Bh + (size_t)s * DIM + f);

        float x0 = ce.x + dh.x + eh.x;
        float x1 = ce.y + dh.y + eh.y;
        float x2 = ce.z + dh.z + eh.z;
        float x3 = ce.w + dh.w + eh.w;

        float g0 = 1.f / (1.f + __expf(-x0));
        float g1 = 1.f / (1.f + __expf(-x1));
        float g2 = 1.f / (1.f + __expf(-x2));
        float g3 = 1.f / (1.f + __expf(-x3));

        float4 xo; xo.x = x0; xo.y = x1; xo.z = x2; xo.w = x3;
        *(float4*)(eij + (size_t)e * DIM + f) = xo;

        float* np = num + (size_t)d * DIM + f;
        float* dp = den + (size_t)d * DIM + f;
        atomicAdd(np + 0, g0 * bh.x);
        atomicAdd(np + 1, g1 * bh.y);
        atomicAdd(np + 2, g2 * bh.z);
        atomicAdd(np + 3, g3 * bh.w);
        atomicAdd(dp + 0, g0);
        atomicAdd(dp + 1, g1);
        atomicAdd(dp + 2, g2);
        atomicAdd(dp + 3, g3);

        ls0 += x0; ls1 += x1; ls2 += x2; ls3 += x3;
        lq0 += x0 * x0; lq1 += x1 * x1; lq2 += x2 * x2; lq3 += x3 * x3;
    }

    atomicAdd(&sSum[f + 0], ls0); atomicAdd(&sSum[f + 1], ls1);
    atomicAdd(&sSum[f + 2], ls2); atomicAdd(&sSum[f + 3], ls3);
    atomicAdd(&sSq[f + 0], lq0);  atomicAdd(&sSq[f + 1], lq1);
    atomicAdd(&sSq[f + 2], lq2);  atomicAdd(&sSq[f + 3], lq3);
    __syncthreads();
    for (int i = tid; i < DIM; i += 256) {
        atomicAdd(&e_sum[i], sSum[i]);
        atomicAdd(&e_sq[i],  sSq[i]);
    }
}

// ---------------------------------------------------------------------------
// h_new = Ah + num/(den+1e-6); store into h_out region; accumulate h BN stats.
// ---------------------------------------------------------------------------
__global__ __launch_bounds__(256)
void node_kernel(const float* __restrict__ Ah, const float* __restrict__ num,
                 const float* __restrict__ den, float* __restrict__ hout,
                 float* __restrict__ h_sum, float* __restrict__ h_sq) {
    __shared__ float sSum[DIM];
    __shared__ float sSq[DIM];
    int tid = threadIdx.x;
    for (int i = tid; i < DIM; i += 256) { sSum[i] = 0.f; sSq[i] = 0.f; }
    __syncthreads();

    int wave = tid >> 5, lane = tid & 31;
    int f = lane * 4;
    long gw     = (long)blockIdx.x * 8 + wave;
    long stride = (long)gridDim.x * 8;

    float ls0 = 0.f, ls1 = 0.f, ls2 = 0.f, ls3 = 0.f;
    float lq0 = 0.f, lq1 = 0.f, lq2 = 0.f, lq3 = 0.f;

    for (long n = gw; n < NN_NODES; n += stride) {
        float4 av = *(const float4*)(Ah  + (size_t)n * DIM + f);
        float4 nv = *(const float4*)(num + (size_t)n * DIM + f);
        float4 dv = *(const float4*)(den + (size_t)n * DIM + f);
        float x0 = av.x + nv.x / (dv.x + 1e-6f);
        float x1 = av.y + nv.y / (dv.y + 1e-6f);
        float x2 = av.z + nv.z / (dv.z + 1e-6f);
        float x3 = av.w + nv.w / (dv.w + 1e-6f);
        float4 xo; xo.x = x0; xo.y = x1; xo.z = x2; xo.w = x3;
        *(float4*)(hout + (size_t)n * DIM + f) = xo;
        ls0 += x0; ls1 += x1; ls2 += x2; ls3 += x3;
        lq0 += x0 * x0; lq1 += x1 * x1; lq2 += x2 * x2; lq3 += x3 * x3;
    }

    atomicAdd(&sSum[f + 0], ls0); atomicAdd(&sSum[f + 1], ls1);
    atomicAdd(&sSum[f + 2], ls2); atomicAdd(&sSum[f + 3], ls3);
    atomicAdd(&sSq[f + 0], lq0);  atomicAdd(&sSq[f + 1], lq1);
    atomicAdd(&sSq[f + 2], lq2);  atomicAdd(&sSq[f + 3], lq3);
    __syncthreads();
    for (int i = tid; i < DIM; i += 256) {
        atomicAdd(&h_sum[i], sSum[i]);
        atomicAdd(&h_sq[i],  sSq[i]);
    }
}

// ---------------------------------------------------------------------------
// Per-feature BN scale/shift from accumulated sums (biased variance).
// ---------------------------------------------------------------------------
__global__ void finalize_kernel(const float* __restrict__ h_sum, const float* __restrict__ h_sq,
                                const float* __restrict__ e_sum, const float* __restrict__ e_sq,
                                const float* __restrict__ gh, const float* __restrict__ bh,
                                const float* __restrict__ ge, const float* __restrict__ be,
                                float* __restrict__ h_scale, float* __restrict__ h_shift,
                                float* __restrict__ e_scale, float* __restrict__ e_shift) {
    int i = threadIdx.x;
    if (i < DIM) {
        float mh = h_sum[i] * (1.f / (float)NN_NODES);
        float vh = h_sq[i]  * (1.f / (float)NN_NODES) - mh * mh;
        float sc = gh[i] * rsqrtf(vh + 1e-5f);
        h_scale[i] = sc;
        h_shift[i] = bh[i] - mh * sc;

        float me  = e_sum[i] * (1.f / (float)NE_EDGES);
        float ve  = e_sq[i]  * (1.f / (float)NE_EDGES) - me * me;
        float sce = ge[i] * rsqrtf(ve + 1e-5f);
        e_scale[i] = sce;
        e_shift[i] = be[i] - me * sce;
    }
}

// ---------------------------------------------------------------------------
// In-place y = relu(x*scale + shift) over [rows x 128]
// ---------------------------------------------------------------------------
__global__ __launch_bounds__(256)
void bn_relu_kernel(float* __restrict__ data, const float* __restrict__ scale,
                    const float* __restrict__ shift, long rows) {
    int wave = threadIdx.x >> 5, lane = threadIdx.x & 31;
    int f = lane * 4;
    long gw     = (long)blockIdx.x * 8 + wave;
    long stride = (long)gridDim.x * 8;
    float4 sc = *(const float4*)(scale + f);
    float4 sh = *(const float4*)(shift + f);
    for (long r = gw; r < rows; r += stride) {
        float4 x = *(float4*)(data + (size_t)r * DIM + f);
        x.x = fmaxf(0.f, x.x * sc.x + sh.x);
        x.y = fmaxf(0.f, x.y * sc.y + sh.y);
        x.z = fmaxf(0.f, x.z * sc.z + sh.z);
        x.w = fmaxf(0.f, x.w * sc.w + sh.w);
        *(float4*)(data + (size_t)r * DIM + f) = x;
    }
}

// ---------------------------------------------------------------------------
extern "C" void kernel_launch(void* const* d_in, const int* in_sizes, int n_in,
                              void* d_out, int out_size, void* d_ws, size_t ws_size,
                              hipStream_t stream) {
    const float* h  = (const float*)d_in[0];
    const float* e  = (const float*)d_in[1];
    const int* src  = (const int*)d_in[2];
    const int* dst  = (const int*)d_in[3];
    const float* WA = (const float*)d_in[4],  *bA = (const float*)d_in[5];
    const float* WB = (const float*)d_in[6],  *bB = (const float*)d_in[7];
    const float* WC = (const float*)d_in[8],  *bC = (const float*)d_in[9];
    const float* WD = (const float*)d_in[10], *bD = (const float*)d_in[11];
    const float* WE = (const float*)d_in[12], *bE = (const float*)d_in[13];
    const float* gamma_h = (const float*)d_in[14], *beta_h = (const float*)d_in[15];
    const float* gamma_e = (const float*)d_in[16], *beta_e = (const float*)d_in[17];

    // ---- workspace layout ----
    char* ws = (char*)d_ws;
    _Float16* wf16 = (_Float16*)ws;                       // 5 * 128*128 f16 panels (fragment-major)
    size_t off = (size_t)5 * DIM * DIM * sizeof(_Float16);
    auto fall = [&](size_t nfloats) {
        float* p = (float*)(ws + off);
        off += nfloats * sizeof(float);
        return p;
    };
    float* Ah  = fall((size_t)NN_NODES * DIM);
    float* Bh  = fall((size_t)NN_NODES * DIM);
    float* Dh  = fall((size_t)NN_NODES * DIM);
    float* Eh  = fall((size_t)NN_NODES * DIM);
    float* Ce  = fall((size_t)NE_EDGES * DIM);
    float* num = fall((size_t)NN_NODES * DIM);
    float* den = fall((size_t)NN_NODES * DIM);
    float* h_sum = fall(DIM);   // next 4 arrays contiguous -> single memset
    float* h_sq  = fall(DIM);
    float* e_sum = fall(DIM);
    float* e_sq  = fall(DIM);
    float* h_scale = fall(DIM);
    float* h_shift = fall(DIM);
    float* e_scale = fall(DIM);
    float* e_shift = fall(DIM);

    float* h_out = (float*)d_out;                          // [50000 x 128]
    float* e_out = h_out + (size_t)NN_NODES * DIM;         // [800000 x 128]

    hipMemsetAsync(num,   0, (size_t)NN_NODES * DIM * sizeof(float), stream);
    hipMemsetAsync(den,   0, (size_t)NN_NODES * DIM * sizeof(float), stream);
    hipMemsetAsync(h_sum, 0, (size_t)4 * DIM * sizeof(float), stream);

    // weight conversion + fragment swizzle (slots: 0=WA 1=WB 2=WC 3=WD 4=WE)
    int cvb = (DIM * DIM + 255) / 256;
    cvt_w_swizzle_kernel<<<cvb, 256, 0, stream>>>(WA, wf16 + 0 * DIM * DIM);
    cvt_w_swizzle_kernel<<<cvb, 256, 0, stream>>>(WB, wf16 + 1 * DIM * DIM);
    cvt_w_swizzle_kernel<<<cvb, 256, 0, stream>>>(WC, wf16 + 2 * DIM * DIM);
    cvt_w_swizzle_kernel<<<cvb, 256, 0, stream>>>(WD, wf16 + 3 * DIM * DIM);
    cvt_w_swizzle_kernel<<<cvb, 256, 0, stream>>>(WE, wf16 + 4 * DIM * DIM);

    // GEMMs
    int nblk = (NN_NODES + 127) / 128;
    int eblk = (NE_EDGES + 127) / 128;
    gemm_wmma_kernel<<<nblk, 256, 0, stream>>>(h, wf16 + 0 * DIM * DIM, bA, Ah, NN_NODES);
    gemm_wmma_kernel<<<nblk, 256, 0, stream>>>(h, wf16 + 1 * DIM * DIM, bB, Bh, NN_NODES);
    gemm_wmma_kernel<<<nblk, 256, 0, stream>>>(h, wf16 + 3 * DIM * DIM, bD, Dh, NN_NODES);
    gemm_wmma_kernel<<<nblk, 256, 0, stream>>>(h, wf16 + 4 * DIM * DIM, bE, Eh, NN_NODES);
    gemm_wmma_kernel<<<eblk, 256, 0, stream>>>(e, wf16 + 2 * DIM * DIM, bC, Ce, NE_EDGES);

    // edge gather/scatter + e BN stats (writes e_ij into d_out e-region)
    edge_kernel<<<2048, 256, 0, stream>>>(Ce, Dh, Eh, Bh, src, dst,
                                          e_out, num, den, e_sum, e_sq);

    // node update + h BN stats (writes h_new into d_out h-region)
    node_kernel<<<1024, 256, 0, stream>>>(Ah, num, den, h_out, h_sum, h_sq);

    // BN finalize + in-place normalize/relu
    finalize_kernel<<<1, 128, 0, stream>>>(h_sum, h_sq, e_sum, e_sq,
                                           gamma_h, beta_h, gamma_e, beta_e,
                                           h_scale, h_shift, e_scale, e_shift);
    bn_relu_kernel<<<1024, 256, 0, stream>>>(h_out, h_scale, h_shift, (long)NN_NODES);
    bn_relu_kernel<<<4096, 256, 0, stream>>>(e_out, e_scale, e_shift, (long)NE_EDGES);
}
